// Hard_Model_42528766165551
// MI455X (gfx1250) — matrix operand order repaired and checked by reading, not compiled, not run on hardware
//
#include <hip/hip_runtime.h>
#include <stdint.h>

#define N_ROW 1024
#define N_COL 65536
#define RANK  128
#define BM    128
#define BN    64
#define LDK   (RANK + 8)   // LDS row stride in halfs: 272B -> conflict-free frag reads

typedef __attribute__((ext_vector_type(16))) _Float16 v16h;
typedef __attribute__((ext_vector_type(8)))  _Float16 v8h;
typedef __attribute__((ext_vector_type(8)))  float    v8f;
typedef __attribute__((ext_vector_type(4)))  uint32_t v4u;
typedef __attribute__((ext_vector_type(8)))  uint32_t v8u;

__device__ __constant__ float tri_tab[4][4] = {
    {1.0f,      0.0f, 0.0f, 0.0f},
    {1.0f,      1.0f, 0.0f, 0.0f},
    {0.5f,      1.0f, 0.5f, 0.0f},
    {1.0f/3.0f, 1.0f, 1.0f, 1.0f/3.0f}};

// LDS byte-offset of a __shared__ object (addrspacecast generic->local, take low 32)
#define LDS_OFF(p) ((uint32_t)(uintptr_t)(__attribute__((address_space(3))) void*)(p))

// ---------------------------------------------------------------------------
// Kernel 1: synthesize C (pseudo-Voigt dictionary), store TRANSPOSED as f16:
//   Ct[n][k],  n in [0,65536), k in [0,128).  One thread = one k, 8 columns.
// ---------------------------------------------------------------------------
__global__ __launch_bounds__(256) void synth_kernel(
    const float* __restrict__ X,
    const float* __restrict__ sigma,
    const float* __restrict__ spacing,
    const float* __restrict__ means,
    const float* __restrict__ Nv,
    const int*   __restrict__ multv,
    _Float16*    __restrict__ Ct)
{
    const int tid = threadIdx.x;
    const int k   = tid & (RANK - 1);
    const int n0  = (blockIdx.x * 2 + (tid >> 7)) * 8;

    const float sx = spacing[k];
    const float sp = (sx > 20.0f) ? sx : log1pf(__expf(sx));       // softplus
    const float sig = fmaxf(sigma[k], 1.0f);
    const float nm  = 1.0f / (1.0f + __expf(-Nv[k]));              // sigmoid
    int mult = multv[k];
    if (mult < 1) mult = 1;
    if (mult > 4) mult = 4;
    const bool  even = (mult & 1) == 0;
    const float mean = means[k];
    const int   hidx = (int)(even ? (mean + 0.5f * sp) : mean);
    const float h    = X[hidx];
    const float mf   = (float)mult;
    const float space0 = even ? (-0.5f * mf * sp + 0.5f * sp)
                              : (-0.5f * (mf - 1.0f) * sp);
    const float gl      = 0.5f * sig;
    const float inv_gl  = 1.0f / gl;
    const float lor_amp = 1.0f / (3.14159265358979f * gl);
    const float gg      = sig * 0.42466090014400953f;   // 1/(2*sqrt(2 ln 2))
    const float inv_gg  = 1.0f / gg;
    const float gau_amp = inv_gg * 0.3989422804014327f; // 1/(gg*sqrt(2 pi))

    float w[4], pos[4];
    #pragma unroll
    for (int j = 0; j < 4; ++j) {
        w[j]   = tri_tab[mult - 1][j] * h;
        pos[j] = mean + space0 + (float)j * sp;
    }

    const float tscale = 65536.0f / 65535.0f;           // jnp.linspace(0, n_col, n_col)
    #pragma unroll
    for (int i = 0; i < 8; ++i) {
        const int   n = n0 + i;
        const float t = (float)n * tscale;
        float acc = 0.0f;
        #pragma unroll
        for (int j = 0; j < 4; ++j) {
            const float d   = t - pos[j];
            const float ul  = d * inv_gl;
            const float lor = lor_amp / (1.0f + ul * ul);
            const float ug  = d * inv_gg;
            const float gau = __expf(-0.5f * ug * ug) * gau_amp;
            acc += w[j] * (nm * lor + (1.0f - nm) * gau);
        }
        Ct[(size_t)n * RANK + k] = (_Float16)acc;
    }
}

// ---------------------------------------------------------------------------
// Kernel 2: W f32 -> f16
// ---------------------------------------------------------------------------
__global__ __launch_bounds__(256) void wconv_kernel(
    const float* __restrict__ W, _Float16* __restrict__ Wf)
{
    const int i = blockIdx.x * 256 + threadIdx.x;
    Wf[i] = (_Float16)W[i];
}

// ---------------------------------------------------------------------------
// TDM: DMA a 2D f16 tile (rows x 128) into LDS, inserting a 16B pad after
// every 256B row => LDS row stride 272B == LDK halfs. D# per ISA 8.3/8.4.
// Must be executed by exactly ONE wave (TDM ignores EXEC).
// ---------------------------------------------------------------------------
__device__ __forceinline__ void tdm_load_tile(uint32_t lds_off,
                                              const _Float16* gsrc,
                                              uint32_t rows)
{
    const uint64_t ga = (uint64_t)(uintptr_t)gsrc;
    v4u g0;
    g0[0] = 1u;                                  // count=1, user desc
    g0[1] = lds_off;                             // lds_addr
    g0[2] = (uint32_t)ga;                        // global_addr[31:0]
    g0[3] = (uint32_t)(ga >> 32) | (2u << 30);   // global_addr[56:32] | type=2
    v8u g1;
    g1[0] = (1u << 16)                           // data_size = 2 bytes
          | (1u << 20)                           // pad_enable
          | (5u << 22)                           // pad_interval: 64 DWORDs (256B)
          | (3u << 25);                          // pad_amount:   4 DWORDs (16B)
    g1[1] = (uint32_t)RANK << 16;                // tensor_dim0 = 128
    g1[2] = 0;                                   // tensor_dim1[15:0] = 0 (dim1 = 0x10000)
    g1[3] = 1u | ((uint32_t)RANK << 16);         // tensor_dim1[31:16]=1 | tile_dim0 = 128
    g1[4] = rows & 0xFFFFu;                      // tile_dim1 = rows, tile_dim2 = 0
    g1[5] = RANK;                                // tensor_dim0_stride = 128
    g1[6] = 0;                                   // dim0_stride hi / dim1_stride lo
    g1[7] = 0;
    asm volatile("tensor_load_to_lds %0, %1" :: "s"(g0), "s"(g1) : "memory");
}

// ---------------------------------------------------------------------------
// Kernel 3: out = W @ C via v_wmma_f32_16x16x32_f16.
// Block = 256 threads (8 waves), tile 128(M) x 64(N), K=128 TDM-staged in LDS.
// ---------------------------------------------------------------------------
__global__ __launch_bounds__(256) void gemm_kernel(
    const _Float16* __restrict__ Wf,   // (N_ROW, RANK)
    const _Float16* __restrict__ Ct,   // (N_COL, RANK)
    float*          __restrict__ out)  // (N_ROW, N_COL)
{
    __shared__ __align__(16) _Float16 As[BM][LDK];   // 34,816 B
    __shared__ __align__(16) _Float16 Bs[BN][LDK];   // 17,408 B

    const int tid = threadIdx.x;
    const int n0  = blockIdx.x * BN;
    const int m0  = blockIdx.y * BM;

    // ---- TDM-stage both tiles from wave 0 only ----
    if (tid < 32) {
        tdm_load_tile(LDS_OFF(&As[0][0]), Wf + (size_t)m0 * RANK, BM);
        tdm_load_tile(LDS_OFF(&Bs[0][0]), Ct + (size_t)n0 * RANK, BN);
        __builtin_amdgcn_s_wait_tensorcnt(0);
    }
    __syncthreads();

    const int lane   = tid & 31;
    const int wave   = tid >> 5;
    const int wm     = wave & 3;          // 0..3 -> M subtile of 32
    const int wn     = wave >> 2;         // 0..1 -> N subtile of 32
    const int lane16 = lane & 15;
    const int hi     = lane >> 4;
    const int mbase  = wm * 32;
    const int nbase  = wn * 32;

    v8f acc[2][2] = {};

    #pragma unroll
    for (int ks = 0; ks < RANK / 32; ++ks) {
        const int kk = ks * 32;
        v16h a[2];
        #pragma unroll
        for (int ti = 0; ti < 2; ++ti) {
            const int row = mbase + ti * 16 + lane16;
            const int kh  = kk + hi * 8;
            *((v8h*)&a[ti])     = *(const v8h*)&As[row][kh];
            *((v8h*)&a[ti] + 1) = *(const v8h*)&As[row][kh + 16];
        }
        v16h b[2];
        #pragma unroll
        for (int tj = 0; tj < 2; ++tj) {
            const int row = nbase + tj * 16 + lane16;
            const int kh  = kk + hi * 8;
            *((v8h*)&b[tj])     = *(const v8h*)&Bs[row][kh];
            *((v8h*)&b[tj] + 1) = *(const v8h*)&Bs[row][kh + 16];
        }
        #pragma unroll
        for (int ti = 0; ti < 2; ++ti)
            #pragma unroll
            for (int tj = 0; tj < 2; ++tj)
                acc[ti][tj] = __builtin_amdgcn_wmma_f32_16x16x32_f16(
                    false, a[ti], false, b[tj], (short)0, acc[ti][tj],
                    false, false);
    }

    // ---- non-temporal stores: 256MB result must not evict Ct/Wf from L2 ----
    #pragma unroll
    for (int ti = 0; ti < 2; ++ti) {
        #pragma unroll
        for (int tj = 0; tj < 2; ++tj) {
            const int row0 = m0 + mbase + ti * 16 + hi * 8;
            const int col  = n0 + nbase + tj * 16 + lane16;
            #pragma unroll
            for (int e = 0; e < 8; ++e)
                __builtin_nontemporal_store(
                    acc[ti][tj][e], &out[(size_t)(row0 + e) * N_COL + col]);
        }
    }
}

// ---------------------------------------------------------------------------
extern "C" void kernel_launch(void* const* d_in, const int* in_sizes, int n_in,
                              void* d_out, int out_size, void* d_ws, size_t ws_size,
                              hipStream_t stream) {
    const float* W       = (const float*)d_in[0];
    const float* X       = (const float*)d_in[1];
    const float* sigma   = (const float*)d_in[2];
    const float* spacing = (const float*)d_in[3];
    const float* means   = (const float*)d_in[4];
    const float* Nv      = (const float*)d_in[5];
    const int*   multv   = (const int*)d_in[6];
    float*       out     = (float*)d_out;

    _Float16* Ct = (_Float16*)d_ws;                                  // 16 MiB
    _Float16* Wf = (_Float16*)((char*)d_ws + (size_t)N_COL * RANK * 2);

    synth_kernel<<<N_COL / 16, 256, 0, stream>>>(X, sigma, spacing, means, Nv,
                                                 multv, Ct);
    wconv_kernel<<<(N_ROW * RANK) / 256, 256, 0, stream>>>(W, Wf);
    gemm_kernel<<<dim3(N_COL / BN, N_ROW / BM), 256, 0, stream>>>(Wf, Ct, out);
}
